// SlidingRNN_v3_27676769255802
// MI455X (gfx1250) — compile-verified, matrix operand order checked by reading, not run on hardware
//
#include <hip/hip_runtime.h>
#include <hip/hip_bf16.h>
#include <math.h>

// ---------------------------------------------------------------------------
// Sizes from the reference: B=64, T=50, C=51, H=256, S=51, M=8
// ---------------------------------------------------------------------------
#define BB 64
#define TT 50
#define CC 51
#define HH 256
#define G3 768          // 3*H
#define SS 51
#define MM 8
#define NT_G 48         // G3 / 16  (WMMA n-tiles of the gate dimension)
#define KT_H 8          // H / 32   (WMMA k-tiles over hidden dim)
#define KT_C 2          // ceil(51/32) -> K padded to 64

typedef __attribute__((ext_vector_type(16))) _Float16 v16h;
typedef __attribute__((ext_vector_type(8)))  float    v8f;

static __device__ __forceinline__ v8f wmma32f16(v16h a, v16h b, v8f c) {
  // D = A(16x32 f16) x B(32x16 f16) + C(16x16 f32)
  return __builtin_amdgcn_wmma_f32_16x16x32_f16(
      /*neg_a=*/false, a, /*neg_b=*/false, b,
      /*c_mod=*/(short)0, c, /*reuse_a=*/false, /*reuse_b=*/false);
}

static __device__ __forceinline__ unsigned pack2h(float a, float b) {
  union { _Float16 h[2]; unsigned u; } u_;
  u_.h[0] = (_Float16)a; u_.h[1] = (_Float16)b;
  return u_.u;
}

static __device__ __forceinline__ float sigmf(float x) {
  return 1.0f / (1.0f + __expf(-x));
}

// Write 8 consecutive hidden-dim values (k0..k0+7, k0 multiple of 8) of row m_l
// into a 16x32 f16 WMMA A-operand tile stored as [lane][8 dwords].
// A layout (ISA 7.12.2): lane<16 -> M=lane, K in {0..7,16..23};
//                        lane>=16 -> M=lane-16, K in {8..15,24..31}.
static __device__ __forceinline__ void write_tileA(unsigned* tileBase, int m_l,
                                                   int kl0, const float* h8) {
  const int kb   = ((kl0 & 15) >= 8) ? 1 : 0;
  const int lane = m_l + (kb ? 16 : 0);
  const int hi   = kl0 >> 4;
  unsigned* d = tileBase + lane * 8 + hi * 4;
#pragma unroll
  for (int jj = 0; jj < 4; ++jj) d[jj] = pack2h(h8[2 * jj], h8[2 * jj + 1]);
}

// ---------------------------------------------------------------------------
// Weight swizzle: W (Grows x Kcols, f32) -> f16 B-operand tiles
// dst layout: [nt][kt][lane][8 dwords], B[k][n] = W[n][k], zero padded.
// ---------------------------------------------------------------------------
__global__ void k_swz(const float* __restrict__ W, int Grows, int Kcols,
                      unsigned* __restrict__ dst, int NT, int KT) {
  int idx = blockIdx.x * blockDim.x + threadIdx.x;
  int total = NT * KT * 32;
  if (idx >= total) return;
  int lane = idx & 31;
  int nt = (idx >> 5) / KT;
  int kt = (idx >> 5) % KT;
  int n  = nt * 16 + (lane & 15);
  int kb = (lane >> 4) * 8;
  unsigned o[8];
#pragma unroll
  for (int v = 0; v < 8; ++v) {
    int k0 = kt * 32 + ((v >> 2) << 4) + kb + ((v & 3) << 1);
    float f0 = (n < Grows && (k0    ) < Kcols) ? W[(size_t)n * Kcols + k0    ] : 0.f;
    float f1 = (n < Grows && (k0 + 1) < Kcols) ? W[(size_t)n * Kcols + k0 + 1] : 0.f;
    o[v] = pack2h(f0, f1);
  }
  unsigned* d = dst + (size_t)idx * 8;
#pragma unroll
  for (int v = 0; v < 8; ++v) d[v] = o[v];
}

// ---------------------------------------------------------------------------
// Copy x (B,T,C) into circular frame buffer frames[f][b][c], f = 0..49
// ---------------------------------------------------------------------------
__global__ void k_frames_init(const float* __restrict__ x,
                              float* __restrict__ frames) {
  int i = blockIdx.x * blockDim.x + threadIdx.x;
  if (i >= TT * BB * CC) return;
  int c = i % CC;
  int r = i / CC;
  int b = r % BB;
  int f = r / BB;
  frames[i] = x[((size_t)b * TT + f) * CC + c];
}

// ---------------------------------------------------------------------------
// feat[f] = frames[f] @ wih0^T + bih0   (64 x 51 -> 64 x 768), WMMA f16
// One block per frame; 8 waves, 24 tiles each (4 mt x 48 nt, K padded to 64).
// ---------------------------------------------------------------------------
__global__ __launch_bounds__(256) void k_feat(
    const float* __restrict__ frames, int f0,
    const unsigned* __restrict__ wih0_sw, const float* __restrict__ bih0,
    float* __restrict__ feat) {
  __shared__ __align__(32) unsigned Alds[4 * KT_C * 32 * 8];
  const int f = f0 + blockIdx.x;
  const int tid  = threadIdx.x;
  const int lane = tid & 31;
  const int wave = tid >> 5;
  {
    int m  = tid >> 2;          // 0..63
    int k0 = (tid & 3) * 16;    // 0,16,32,48
    const float* src = frames + ((size_t)f * BB + m) * CC;
    float v[16];
#pragma unroll
    for (int j = 0; j < 16; ++j) { int c = k0 + j; v[j] = (c < CC) ? src[c] : 0.f; }
    int mt = m >> 4, ml = m & 15;
    int kt = k0 >> 5;
    unsigned* tb = &Alds[((mt * KT_C) + kt) * 256];
    write_tileA(tb, ml, (k0    ) & 31, v    );
    write_tileA(tb, ml, (k0 + 8) & 31, v + 8);
  }
  __syncthreads();
  float* drow = feat + (size_t)f * BB * G3;
  for (int i = 0; i < 24; ++i) {
    int tile = wave + i * 8;    // 0..191
    int mt = tile / NT_G;
    int nt = tile % NT_G;
    v8f acc = {0.f,0.f,0.f,0.f,0.f,0.f,0.f,0.f};
#pragma unroll
    for (int kt = 0; kt < KT_C; ++kt) {
      v16h a = *(const v16h*)&Alds[((mt * KT_C) + kt) * 256 + lane * 8];
      v16h b = *(const v16h*)&wih0_sw[(((size_t)nt * KT_C) + kt) * 256 + lane * 8];
      acc = wmma32f16(a, b, acc);
    }
    int n0 = nt * 16 + (lane & 15);
    int m0 = mt * 16 + (lane >> 4) * 8;
    float bb = bih0[n0];
#pragma unroll
    for (int a8 = 0; a8 < 8; ++a8)
      drow[(size_t)(m0 + a8) * G3 + n0] = acc[a8] + bb;
  }
}

// ---------------------------------------------------------------------------
// K2: the sequential core. One workgroup (32 waves) runs the full 50-step
// dual-layer GRU scan; hidden state kept in registers + LDS (WMMA A layout,
// f16). Two row-passes of 32 keep LDS at 128KB. Ends with dec = h1@wdec^T.
// ---------------------------------------------------------------------------
__global__ __launch_bounds__(1024) void k2_recurrent(
    const float* __restrict__ feat, int t,
    const unsigned* __restrict__ whh0_sw,
    const unsigned* __restrict__ wih1_sw,
    const unsigned* __restrict__ whh1_sw,
    const unsigned* __restrict__ wdec_sw,
    const float* __restrict__ bhh0, const float* __restrict__ bih1,
    const float* __restrict__ bhh1, const float* __restrict__ bdec,
    float* __restrict__ decbuf) {
  __shared__ __align__(32) _Float16 Gs[32 * G3];    // 48 KB: gate pre-activations
  __shared__ __align__(32) _Float16 GN[32 * HH];    // 16 KB: gh1 n-section alone
  __shared__ __align__(32) unsigned h0A[4 * KT_H * 32 * 8];  // 32 KB f16 A-layout
  __shared__ __align__(32) unsigned h1A[4 * KT_H * 32 * 8];  // 32 KB

  const int tid  = threadIdx.x;
  const int lane = tid & 31;
  const int wave = tid >> 5;

  for (int i = tid; i < 4 * KT_H * 32 * 8; i += 1024) { h0A[i] = 0u; h1A[i] = 0u; }
  float h0r[2][8], h1r[2][8];
#pragma unroll
  for (int p = 0; p < 2; ++p)
#pragma unroll
    for (int j = 0; j < 8; ++j) { h0r[p][j] = 0.f; h1r[p][j] = 0.f; }
  __syncthreads();

  const int upd_m  = wave;          // row within the 32-row pass block
  const int upd_k0 = lane * 8;      // 8 consecutive hidden cols per thread
  const int upd_kt = upd_k0 >> 5;
  const int upd_kl = upd_k0 & 31;

  for (int tt = 0; tt < TT; ++tt) {
    const float* featRowBase = feat + ((size_t)(t + tt)) * BB * G3;
#pragma unroll
    for (int p = 0; p < 2; ++p) {
      // ---- phase A: gh0 = h0 @ whh0^T for rows [p*32, p*32+32) ----
#pragma unroll
      for (int ti = 0; ti < 3; ++ti) {
        int tile = wave + ti * 32;   // 96 tiles = 2 mtL x 48 nt
        int mtL = tile / NT_G;
        int nt  = tile % NT_G;
        int mt  = p * 2 + mtL;
        v8f acc = {0.f,0.f,0.f,0.f,0.f,0.f,0.f,0.f};
#pragma unroll
        for (int kt = 0; kt < KT_H; ++kt) {
          v16h a = *(const v16h*)&h0A[((mt * KT_H) + kt) * 256 + lane * 8];
          v16h b = *(const v16h*)&whh0_sw[(((size_t)nt * KT_H) + kt) * 256 + lane * 8];
          acc = wmma32f16(a, b, acc);
        }
        int n0 = nt * 16 + (lane & 15);
        int mr = mtL * 16 + (lane >> 4) * 8;
        float bb = bhh0[n0];
#pragma unroll
        for (int a8 = 0; a8 < 8; ++a8)
          Gs[(mr + a8) * G3 + n0] = (_Float16)(acc[a8] + bb);
      }
      __syncthreads();
      // ---- h0 gate update ----
      {
        int m = p * 32 + upd_m;
        const float* fr = featRowBase + (size_t)m * G3;
        float hnew[8];
#pragma unroll
        for (int j = 0; j < 8; ++j) {
          int k = upd_k0 + j;
          float r  = sigmf(fr[k]        + (float)Gs[upd_m * G3 + k]);
          float z  = sigmf(fr[HH + k]   + (float)Gs[upd_m * G3 + HH + k]);
          float hn = (float)Gs[upd_m * G3 + 2 * HH + k];
          float nn = tanhf(fr[2 * HH + k] + r * hn);
          hnew[j] = (1.f - z) * nn + z * h0r[p][j];
          h0r[p][j] = hnew[j];
        }
        write_tileA(&h0A[(((m >> 4) * KT_H) + upd_kt) * 256], m & 15, upd_kl, hnew);
      }
      __syncthreads();
      // ---- phase B: u1 = h0_new@wih1^T and gh1 = h1@whh1^T (fused) ----
#pragma unroll
      for (int ti = 0; ti < 3; ++ti) {
        int tile = wave + ti * 32;
        int mtL = tile / NT_G;
        int nt  = tile % NT_G;
        int mt  = p * 2 + mtL;
        v8f accU = {0.f,0.f,0.f,0.f,0.f,0.f,0.f,0.f};
        v8f accG = {0.f,0.f,0.f,0.f,0.f,0.f,0.f,0.f};
#pragma unroll
        for (int kt = 0; kt < KT_H; ++kt) {
          v16h a0 = *(const v16h*)&h0A[((mt * KT_H) + kt) * 256 + lane * 8];
          v16h a1 = *(const v16h*)&h1A[((mt * KT_H) + kt) * 256 + lane * 8];
          v16h b0 = *(const v16h*)&wih1_sw[(((size_t)nt * KT_H) + kt) * 256 + lane * 8];
          v16h b1 = *(const v16h*)&whh1_sw[(((size_t)nt * KT_H) + kt) * 256 + lane * 8];
          accU = wmma32f16(a0, b0, accU);
          accG = wmma32f16(a1, b1, accG);
        }
        int n0 = nt * 16 + (lane & 15);
        int mr = mtL * 16 + (lane >> 4) * 8;
        float bi = bih1[n0], bh = bhh1[n0];
#pragma unroll
        for (int a8 = 0; a8 < 8; ++a8) {
          float gval = accG[a8] + bh;
          Gs[(mr + a8) * G3 + n0] = (_Float16)(accU[a8] + bi + gval);
          if (nt >= 32) GN[(mr + a8) * HH + (n0 - 2 * HH)] = (_Float16)gval;
        }
      }
      __syncthreads();
      // ---- h1 gate update ----
      {
        int m = p * 32 + upd_m;
        float hnew[8];
#pragma unroll
        for (int j = 0; j < 8; ++j) {
          int k = upd_k0 + j;
          float r  = sigmf((float)Gs[upd_m * G3 + k]);
          float z  = sigmf((float)Gs[upd_m * G3 + HH + k]);
          float hn = (float)GN[upd_m * HH + k];
          float xn = (float)Gs[upd_m * G3 + 2 * HH + k] - hn;  // recover xn
          float nn = tanhf(xn + r * hn);
          hnew[j] = (1.f - z) * nn + z * h1r[p][j];
          h1r[p][j] = hnew[j];
        }
        write_tileA(&h1A[(((m >> 4) * KT_H) + upd_kt) * 256], m & 15, upd_kl, hnew);
      }
      __syncthreads();
    }
  }
  // ---- dec = h1_T @ wdec^T + bdec : 4 mt x 4 nt tiles, waves 0..15 ----
  if (wave < 16) {
    int mt = wave >> 2;
    int nt = wave & 3;
    v8f acc = {0.f,0.f,0.f,0.f,0.f,0.f,0.f,0.f};
#pragma unroll
    for (int kt = 0; kt < KT_H; ++kt) {
      v16h a = *(const v16h*)&h1A[((mt * KT_H) + kt) * 256 + lane * 8];
      v16h b = *(const v16h*)&wdec_sw[(((size_t)nt * KT_H) + kt) * 256 + lane * 8];
      acc = wmma32f16(a, b, acc);
    }
    int n0 = nt * 16 + (lane & 15);
    int m0 = mt * 16 + (lane >> 4) * 8;
    float bb = (n0 < CC) ? bdec[n0] : 0.f;
#pragma unroll
    for (int a8 = 0; a8 < 8; ++a8)
      decbuf[(m0 + a8) * 64 + n0] = acc[a8] + bb;
  }
}

// ---------------------------------------------------------------------------
// K3: mini-MLP mixer + output projection + merge + residual, per batch.
// f (C=51 x S=51) in LDS, 8 iters of (f@wt^T + LN over C + affine + residual),
// then nf[c] = sum_c' w_out[c][c'] * (sum_s f[c'][s]*wm[s]) + bias terms + last.
// ---------------------------------------------------------------------------
__global__ __launch_bounds__(256) void k3_mini(
    float* __restrict__ frames, int t, const float* __restrict__ decbuf,
    const float* __restrict__ wt, const float* __restrict__ bt,
    const float* __restrict__ alpha, const float* __restrict__ beta,
    const float* __restrict__ w_out, const float* __restrict__ b_out,
    const float* __restrict__ w_merge, const float* __restrict__ b_merge,
    float* __restrict__ out) {
  const int b = blockIdx.x;
  const int tid = threadIdx.x;
  __shared__ float F[CC][SS + 1];
  __shared__ float Y[CC][SS + 1];
  __shared__ float mu[SS], rs[SS], rvec[CC];
  __shared__ float wmsum;

  for (int i = tid; i < CC * SS; i += 256) {
    int c = i / SS, s = i % SS;
    F[c][s] = (s < TT) ? frames[(((size_t)(t + s)) * BB + b) * CC + c]
                       : decbuf[b * 64 + c];
  }
  __syncthreads();

  for (int it = 0; it < MM; ++it) {
    const float* Wt = wt + (size_t)it * SS * SS;
    const float* Bt = bt + it * SS;
    for (int i = tid; i < CC * SS; i += 256) {
      int c = i / SS, s = i % SS;
      float acc = Bt[s];
      for (int sp = 0; sp < SS; ++sp) acc += F[c][sp] * Wt[s * SS + sp];
      Y[c][s] = acc;
    }
    __syncthreads();
    if (tid < SS) {
      int s = tid;
      float m = 0.f;
      for (int c = 0; c < CC; ++c) m += Y[c][s];
      m /= (float)CC;
      float v = 0.f;
      for (int c = 0; c < CC; ++c) { float d = Y[c][s] - m; v += d * d; }
      v /= (float)CC;
      mu[s] = m;
      rs[s] = rsqrtf(v + 1e-5f);
    }
    __syncthreads();
    for (int i = tid; i < CC * SS; i += 256) {
      int c = i / SS, s = i % SS;
      F[c][s] += (Y[c][s] - mu[s]) * rs[s] * alpha[it * CC + c] + beta[it * CC + c];
    }
    __syncthreads();
  }

  if (tid < CC) {
    int cp = tid;
    float acc = 0.f;
    for (int s = 0; s < SS; ++s) acc += F[cp][s] * w_merge[s];
    rvec[cp] = acc;
  }
  if (tid == 0) {
    float sm = 0.f;
    for (int s = 0; s < SS; ++s) sm += w_merge[s];
    wmsum = sm;
  }
  __syncthreads();
  if (tid < CC) {
    int c = tid;
    float acc = 0.f;
    for (int cp = 0; cp < CC; ++cp) acc += w_out[c * CC + cp] * rvec[cp];
    float nf = acc + b_out[c] * wmsum + b_merge[0] +
               frames[(((size_t)(TT - 1 + t)) * BB + b) * CC + c];  // + last
    out[((size_t)b * TT + t) * CC + c] = nf;
    frames[(((size_t)(TT + t)) * BB + b) * CC + c] = nf;
  }
}

// ---------------------------------------------------------------------------
// Host launcher
// ---------------------------------------------------------------------------
extern "C" void kernel_launch(void* const* d_in, const int* in_sizes, int n_in,
                              void* d_out, int out_size, void* d_ws, size_t ws_size,
                              hipStream_t stream) {
  const float* x     = (const float*)d_in[0];
  const float* wih0  = (const float*)d_in[1];
  const float* whh0  = (const float*)d_in[2];
  const float* bih0  = (const float*)d_in[3];
  const float* bhh0  = (const float*)d_in[4];
  const float* wih1  = (const float*)d_in[5];
  const float* whh1  = (const float*)d_in[6];
  const float* bih1  = (const float*)d_in[7];
  const float* bhh1  = (const float*)d_in[8];
  const float* wdec  = (const float*)d_in[9];
  const float* bdec  = (const float*)d_in[10];
  const float* wt    = (const float*)d_in[11];
  const float* bt    = (const float*)d_in[12];
  const float* alpha = (const float*)d_in[13];
  const float* beta  = (const float*)d_in[14];
  const float* w_out = (const float*)d_in[15];
  const float* b_out = (const float*)d_in[16];
  const float* w_mrg = (const float*)d_in[17];
  const float* b_mrg = (const float*)d_in[18];
  float* out = (float*)d_out;
  char* ws = (char*)d_ws;

  const size_t SZ_FRAMES = (size_t)100 * BB * CC * sizeof(float);      // 1.3 MB
  const size_t SZ_FEAT   = (size_t)100 * BB * G3 * sizeof(float);      // 19.7 MB
  const size_t SZ_DEC    = (size_t)64 * 64 * sizeof(float);
  const size_t SZ_WIH0   = (size_t)NT_G * KT_C * 32 * 8 * 4;
  const size_t SZ_WHH    = (size_t)NT_G * KT_H * 32 * 8 * 4;
  const size_t SZ_WDEC   = (size_t)4 * KT_H * 32 * 8 * 4;

  size_t off = 0;
  float*    frames  = (float*)(ws + off);    off += SZ_FRAMES;
  float*    feat    = (float*)(ws + off);    off += SZ_FEAT;
  float*    decbuf  = (float*)(ws + off);    off += SZ_DEC;
  unsigned* wih0_sw = (unsigned*)(ws + off); off += SZ_WIH0;
  unsigned* whh0_sw = (unsigned*)(ws + off); off += SZ_WHH;
  unsigned* wih1_sw = (unsigned*)(ws + off); off += SZ_WHH;
  unsigned* whh1_sw = (unsigned*)(ws + off); off += SZ_WHH;
  unsigned* wdec_sw = (unsigned*)(ws + off); off += SZ_WDEC;
  (void)ws_size; (void)in_sizes; (void)n_in; (void)out_size;

  // --- one-time prep (deterministic, re-run every call) ---
  k_frames_init<<<(TT * BB * CC + 255) / 256, 256, 0, stream>>>(x, frames);
  k_swz<<<(NT_G * KT_C * 32 + 255) / 256, 256, 0, stream>>>(wih0, G3, CC,  wih0_sw, NT_G, KT_C);
  k_swz<<<(NT_G * KT_H * 32 + 255) / 256, 256, 0, stream>>>(whh0, G3, HH,  whh0_sw, NT_G, KT_H);
  k_swz<<<(NT_G * KT_H * 32 + 255) / 256, 256, 0, stream>>>(wih1, G3, HH,  wih1_sw, NT_G, KT_H);
  k_swz<<<(NT_G * KT_H * 32 + 255) / 256, 256, 0, stream>>>(whh1, G3, HH,  whh1_sw, NT_G, KT_H);
  k_swz<<<(4 * KT_H * 32 + 255) / 256, 256, 0, stream>>>(wdec, CC, HH, wdec_sw, 4, KT_H);
  k_feat<<<TT, 256, 0, stream>>>(frames, 0, wih0_sw, bih0, feat);   // frames 0..49

  // --- outer recurrence ---
  for (int t = 0; t < TT; ++t) {
    k2_recurrent<<<1, 1024, 0, stream>>>(feat, t, whh0_sw, wih1_sw, whh1_sw,
                                         wdec_sw, bhh0, bih1, bhh1, bdec, decbuf);
    k3_mini<<<BB, 256, 0, stream>>>(frames, t, decbuf, wt, bt, alpha, beta,
                                    w_out, b_out, w_mrg, b_mrg, out);
    k_feat<<<1, 256, 0, stream>>>(frames, TT + t, wih0_sw, bih0, feat);
  }
}
